// AttentionDecoder_72559177499313
// MI455X (gfx1250) — matrix operand order repaired and checked by reading, not compiled, not run on hardware
//
#include <hip/hip_runtime.h>

// ---------------------------------------------------------------------------
// Problem constants (from reference)
// ---------------------------------------------------------------------------
#define VSZ 32000
#define ESZ 512
#define HSZ 1024
#define LSZ 2
#define BSZ 32
#define TSZ 32
#define S1SZ 64
#define S2SZ 128
#define G3H (3 * HSZ)   // 3072
#define PSZ (ESZ + HSZ) // 1536  (pred width)

typedef __bf16 v16bf __attribute__((ext_vector_type(16)));
typedef __bf16 v8bf  __attribute__((ext_vector_type(8)));
typedef float  v8f   __attribute__((ext_vector_type(8)));

__device__ __forceinline__ __bf16 f2bf(float f) {
    // round-to-nearest-even fp32 -> bf16
    unsigned u = __builtin_bit_cast(unsigned, f);
    unsigned r = (u + 0x7FFFu + ((u >> 16) & 1u)) >> 16;
    unsigned short s = (unsigned short)r;
    return __builtin_bit_cast(__bf16, s);
}

// ---------------------------------------------------------------------------
// fp32 -> bf16 conversion (weights, once per launch)
// ---------------------------------------------------------------------------
__global__ void k_f32_to_bf16(const float* __restrict__ src, __bf16* __restrict__ dst, int n) {
    for (int i = blockIdx.x * blockDim.x + threadIdx.x; i < n; i += gridDim.x * blockDim.x)
        dst[i] = f2bf(src[i]);
}

// zero h0/h1 (f32 + bf16)
__global__ void k_init_hidden(float* h0f, float* h1f, __bf16* h0b, __bf16* h1b) {
    int i = blockIdx.x * blockDim.x + threadIdx.x;
    if (i < BSZ * HSZ) { h0f[i] = 0.f; h1f[i] = 0.f; h0b[i] = f2bf(0.f); h1b[i] = f2bf(0.f); }
}

// ---------------------------------------------------------------------------
// Embedding gather -> bf16  emb[b,e] = embedding[seq[b,t], e]
// ---------------------------------------------------------------------------
__global__ void k_embed(const int* __restrict__ seq, const float* __restrict__ emb_tab,
                        __bf16* __restrict__ out, int t) {
    int i = blockIdx.x * blockDim.x + threadIdx.x; // over B*E
    if (i >= BSZ * ESZ) return;
    int b = i / ESZ, e = i % ESZ;
    int tok = seq[b * TSZ + t];
    out[i] = f2bf(emb_tab[(size_t)tok * ESZ + e]);
}

// ---------------------------------------------------------------------------
// WMMA bf16 GEMM:  C[32,N] = A[32,K] @ W[N,K]^T (+bias)
// 128-thread blocks (4 waves). Each wave computes a 32x64 macro-tile:
// 2 M-tiles x 4 N-tiles; A fragments are loaded once per k-step and
// reused across 8 WMMAs (cuts activation refetch traffic 4x).
// Requires N % 256 == 0 (true for 1024 / 3072 / 32000) and K % 32 == 0.
// ---------------------------------------------------------------------------
__global__ void __launch_bounds__(128)
k_gemm_bf16(const __bf16* __restrict__ A, int lda,
            const __bf16* __restrict__ W, int ldw,
            float* __restrict__ C, int ldc,
            const float* __restrict__ bias, int K) {
    const int wave  = threadIdx.x >> 5;
    const int lane  = threadIdx.x & 31;
    const int sel   = lane >> 4;      // 0 or 1
    const int idx   = lane & 15;
    const int nbase = (blockIdx.x * 4 + wave) * 64; // 4 N-tiles of 16

    const __bf16* a0 = A + (size_t)idx * lda;        // rows 0..15
    const __bf16* a1 = A + (size_t)(16 + idx) * lda; // rows 16..31
    const __bf16* wr0 = W + (size_t)(nbase + 0  + idx) * ldw;
    const __bf16* wr1 = W + (size_t)(nbase + 16 + idx) * ldw;
    const __bf16* wr2 = W + (size_t)(nbase + 32 + idx) * ldw;
    const __bf16* wr3 = W + (size_t)(nbase + 48 + idx) * ldw;

    v8f acc0[4] = {{}, {}, {}, {}};
    v8f acc1[4] = {{}, {}, {}, {}};

    for (int k0 = 0; k0 < K; k0 += 32) {
        // A fragments: lane holds K = {sel*8 .. +7} and {sel*8+16 .. +7}
        v8bf a0lo = *(const v8bf*)(a0 + k0 + sel * 8);
        v8bf a0hi = *(const v8bf*)(a0 + k0 + sel * 8 + 16);
        v8bf a1lo = *(const v8bf*)(a1 + k0 + sel * 8);
        v8bf a1hi = *(const v8bf*)(a1 + k0 + sel * 8 + 16);
        v16bf fa0, fa1;
#pragma unroll
        for (int i = 0; i < 8; ++i) {
            fa0[i] = a0lo[i]; fa0[i + 8] = a0hi[i];
            fa1[i] = a1lo[i]; fa1[i + 8] = a1hi[i];
        }

        const __bf16* wrs[4] = {wr0, wr1, wr2, wr3};
#pragma unroll
        for (int tIdx = 0; tIdx < 4; ++tIdx) {
            // B fragment: lane holds K = {sel*8 .. +7} and {sel*8+16 .. +7}
            v8bf blo = *(const v8bf*)(wrs[tIdx] + k0 + sel * 8);
            v8bf bhi = *(const v8bf*)(wrs[tIdx] + k0 + sel * 8 + 16);
            v16bf fb;
#pragma unroll
            for (int i = 0; i < 8; ++i) { fb[i] = blo[i]; fb[i + 8] = bhi[i]; }
            acc0[tIdx] = __builtin_amdgcn_wmma_f32_16x16x32_bf16(false, fa0, false, fb, (short)0, acc0[tIdx], false, false);
            acc1[tIdx] = __builtin_amdgcn_wmma_f32_16x16x32_bf16(false, fa1, false, fb, (short)0, acc1[tIdx], false, false);
        }
    }

#pragma unroll
    for (int tIdx = 0; tIdx < 4; ++tIdx) {
        const int col = nbase + tIdx * 16 + idx;
        const float bv = bias ? bias[col] : 0.f;
#pragma unroll
        for (int r = 0; r < 8; ++r) {
            int row = r + sel * 8;                    // D layout: VGPR r -> M = r + sel*8
            C[(size_t)row * ldc + col]        = acc0[tIdx][r] + bv;
            C[(size_t)(row + 16) * ldc + col] = acc1[tIdx][r] + bv;
        }
    }
}

// ---------------------------------------------------------------------------
// GRU gate nonlinearity:  h = (1-z)*n + z*h_prev
// ---------------------------------------------------------------------------
__global__ void k_gru_gate(const float* __restrict__ gi, const float* __restrict__ gh,
                           const float* __restrict__ bih, const float* __restrict__ bhh,
                           const float* __restrict__ hprev,
                           float* __restrict__ hout, __bf16* __restrict__ hout_bf) {
    int i = blockIdx.x * blockDim.x + threadIdx.x; // over B*H
    if (i >= BSZ * HSZ) return;
    int b = i / HSZ, j = i % HSZ;
    const float* gib = gi + (size_t)b * G3H;
    const float* ghb = gh + (size_t)b * G3H;
    float ir = gib[j] + bih[j],                 hr = ghb[j] + bhh[j];
    float iz = gib[HSZ + j] + bih[HSZ + j],     hz = ghb[HSZ + j] + bhh[HSZ + j];
    float in = gib[2*HSZ + j] + bih[2*HSZ + j], hn = ghb[2*HSZ + j] + bhh[2*HSZ + j];
    float r = 1.f / (1.f + __expf(-(ir + hr)));
    float z = 1.f / (1.f + __expf(-(iz + hz)));
    float n = tanhf(in + r * hn);
    float h = (1.f - z) * n + z * hprev[i];
    hout[i] = h;
    hout_bf[i] = f2bf(h);
}

// ---------------------------------------------------------------------------
// Attention scores: sc[b,s] = dot(ctx[b,s,:], tq[b,:])   (one wave per (b,s))
// ---------------------------------------------------------------------------
__global__ void k_attn_scores(const float* __restrict__ ctx, const float* __restrict__ tq,
                              float* __restrict__ sc, int S) {
    int wave = (blockIdx.x * blockDim.x + threadIdx.x) >> 5;
    int lane = threadIdx.x & 31;
    if (wave >= BSZ * S) return;
    int b = wave / S, s = wave % S;
    const float* c = ctx + ((size_t)b * S + s) * HSZ;
    const float* q = tq + (size_t)b * HSZ;
    float acc = 0.f;
    for (int k = lane; k < HSZ; k += 32) acc += c[k] * q[k];
#pragma unroll
    for (int off = 16; off > 0; off >>= 1) acc += __shfl_down(acc, off, 32);
    if (lane == 0) sc[b * S + s] = acc;
}

// softmax over S with mask; writes weights to ws and to d_out slice
__global__ void k_attn_softmax(const float* __restrict__ sc, const unsigned char* __restrict__ mask,
                               float* __restrict__ a_ws, float* __restrict__ a_out, int S) {
    __shared__ float red[S2SZ];
    int b = blockIdx.x, s = threadIdx.x;
    float v = sc[b * S + s];
    if (mask[b * S + s]) v = -1e9f;
    red[s] = v; __syncthreads();
    for (int off = S >> 1; off > 0; off >>= 1) {
        if (s < off) red[s] = fmaxf(red[s], red[s + off]);
        __syncthreads();
    }
    float mx = red[0]; __syncthreads();
    float e = __expf(v - mx);
    red[s] = e; __syncthreads();
    for (int off = S >> 1; off > 0; off >>= 1) {
        if (s < off) red[s] += red[s + off];
        __syncthreads();
    }
    float w = e / red[0];
    a_ws[b * S + s] = w;
    a_out[b * S + s] = w;
}

// wctx[b,:] = sum_s a[b,s]*ctx[b,s,:]; pack cat = [wctx, h_in] as bf16
__global__ void k_wctx_cat(const float* __restrict__ a, const float* __restrict__ ctx,
                           const float* __restrict__ hin, __bf16* __restrict__ cat, int S) {
    int i = blockIdx.x * blockDim.x + threadIdx.x; // over B*H
    if (i >= BSZ * HSZ) return;
    int b = i / HSZ, j = i % HSZ;
    const float* ab = a + (size_t)b * S;
    const float* cb = ctx + (size_t)b * S * HSZ + j;
    float acc = 0.f;
    for (int s = 0; s < S; ++s) acc += ab[s] * cb[(size_t)s * HSZ];
    cat[(size_t)b * 2 * HSZ + j]       = f2bf(acc);
    cat[(size_t)b * 2 * HSZ + HSZ + j] = f2bf(hin[i]);
}

// tanh -> f32 + bf16
__global__ void k_tanh(const float* __restrict__ x, float* __restrict__ of, __bf16* __restrict__ ob) {
    int i = blockIdx.x * blockDim.x + threadIdx.x;
    if (i >= BSZ * HSZ) return;
    float v = tanhf(x[i]);
    of[i] = v; ob[i] = f2bf(v);
}

// pred = [output, emb] bf16
__global__ void k_pred_cat(const __bf16* __restrict__ outb, const __bf16* __restrict__ embb,
                           __bf16* __restrict__ pred) {
    int i = blockIdx.x * blockDim.x + threadIdx.x; // over B*P
    if (i >= BSZ * PSZ) return;
    int b = i / PSZ, j = i % PSZ;
    pred[i] = (j < HSZ) ? outb[b * HSZ + j] : embb[b * ESZ + (j - HSZ)];
}

// block-per-row log-softmax over V
__global__ void k_log_softmax(const float* __restrict__ x, float* __restrict__ out) {
    __shared__ float red[256];
    int b = blockIdx.x;
    const float* xb = x + (size_t)b * VSZ;
    float* ob = out + (size_t)b * VSZ;
    float mx = -3.4e38f;
    for (int i = threadIdx.x; i < VSZ; i += 256) mx = fmaxf(mx, xb[i]);
    red[threadIdx.x] = mx; __syncthreads();
    for (int off = 128; off > 0; off >>= 1) {
        if ((int)threadIdx.x < off) red[threadIdx.x] = fmaxf(red[threadIdx.x], red[threadIdx.x + off]);
        __syncthreads();
    }
    mx = red[0]; __syncthreads();
    float sm = 0.f;
    for (int i = threadIdx.x; i < VSZ; i += 256) sm += __expf(xb[i] - mx);
    red[threadIdx.x] = sm; __syncthreads();
    for (int off = 128; off > 0; off >>= 1) {
        if ((int)threadIdx.x < off) red[threadIdx.x] += red[threadIdx.x + off];
        __syncthreads();
    }
    float lse = mx + __logf(red[0]);
    for (int i = threadIdx.x; i < VSZ; i += 256) ob[i] = xb[i] - lse;
}

// hidden = stack([h0,h1]); output = att2
__global__ void k_finalize(const float* __restrict__ h0, const float* __restrict__ h1,
                           const float* __restrict__ att2,
                           float* __restrict__ hid_out, float* __restrict__ out_out) {
    int i = blockIdx.x * blockDim.x + threadIdx.x;
    if (i >= BSZ * HSZ) return;
    hid_out[i] = h0[i];
    hid_out[BSZ * HSZ + i] = h1[i];
    out_out[i] = att2[i];
}

// ---------------------------------------------------------------------------
// host
// ---------------------------------------------------------------------------
static inline size_t alignUp64(size_t x) { return (x + 63) & ~(size_t)63; }

extern "C" void kernel_launch(void* const* d_in, const int* in_sizes, int n_in,
                              void* d_out, int out_size, void* d_ws, size_t ws_size,
                              hipStream_t stream) {
    (void)in_sizes; (void)n_in; (void)out_size; (void)ws_size;

    const int*   seq    = (const int*)d_in[0];
    // d_in[1] prev_output: unused (overwritten before use in reference step)
    const float* ctx1   = (const float*)d_in[2];
    const unsigned char* mask1 = (const unsigned char*)d_in[3];
    const float* ctx2   = (const float*)d_in[4];
    const unsigned char* mask2 = (const unsigned char*)d_in[5];
    const float* embTab = (const float*)d_in[6];
    const float* Wih0 = (const float*)d_in[7];
    const float* Whh0 = (const float*)d_in[8];
    const float* bih0 = (const float*)d_in[9];
    const float* bhh0 = (const float*)d_in[10];
    const float* Wih1 = (const float*)d_in[11];
    const float* Whh1 = (const float*)d_in[12];
    const float* bih1 = (const float*)d_in[13];
    const float* bhh1 = (const float*)d_in[14];
    const float* tWin  = (const float*)d_in[15];
    const float* tWout = (const float*)d_in[16];
    const float* sWin  = (const float*)d_in[17];
    const float* sWout = (const float*)d_in[18];
    const float* outW = (const float*)d_in[19];
    const float* outB = (const float*)d_in[20];

    // ---- workspace layout ----
    char* ws = (char*)d_ws;
    size_t cur = 0;
    auto takeBF = [&](size_t n) { size_t o = cur; cur = alignUp64(cur + n * 2); return (__bf16*)(ws + o); };
    auto takeF  = [&](size_t n) { size_t o = cur; cur = alignUp64(cur + n * 4); return (float*)(ws + o); };

    __bf16* wih0b  = takeBF((size_t)G3H * ESZ);
    __bf16* whh0b  = takeBF((size_t)G3H * HSZ);
    __bf16* wih1b  = takeBF((size_t)G3H * HSZ);
    __bf16* whh1b  = takeBF((size_t)G3H * HSZ);
    __bf16* tWinb  = takeBF((size_t)HSZ * HSZ);
    __bf16* tWoutb = takeBF((size_t)HSZ * 2 * HSZ);
    __bf16* sWinb  = takeBF((size_t)HSZ * HSZ);
    __bf16* sWoutb = takeBF((size_t)HSZ * 2 * HSZ);
    __bf16* outWb  = takeBF((size_t)VSZ * PSZ);

    __bf16* embb  = takeBF((size_t)BSZ * ESZ);
    __bf16* h0b   = takeBF((size_t)BSZ * HSZ);
    __bf16* h1b   = takeBF((size_t)BSZ * HSZ);
    __bf16* catb  = takeBF((size_t)BSZ * 2 * HSZ);
    __bf16* predb = takeBF((size_t)BSZ * PSZ);
    __bf16* att1b = takeBF((size_t)BSZ * HSZ);
    __bf16* att2b = takeBF((size_t)BSZ * HSZ);

    float* h0f   = takeF((size_t)BSZ * HSZ);
    float* h1f   = takeF((size_t)BSZ * HSZ);
    float* gi    = takeF((size_t)BSZ * G3H);
    float* gh    = takeF((size_t)BSZ * G3H);
    float* tq    = takeF((size_t)BSZ * HSZ);
    float* sc    = takeF((size_t)BSZ * S2SZ);
    float* aws   = takeF((size_t)BSZ * S2SZ);
    float* pre   = takeF((size_t)BSZ * HSZ);
    float* att1f = takeF((size_t)BSZ * HSZ);
    float* att2f = takeF((size_t)BSZ * HSZ);
    float* logit = takeF((size_t)BSZ * VSZ);

    // ---- output layout ----
    float* o_scores = (float*)d_out;
    float* o_hidden = o_scores + (size_t)TSZ * BSZ * VSZ;
    float* o_output = o_hidden + (size_t)LSZ * BSZ * HSZ;
    float* o_tree   = o_output + (size_t)BSZ * HSZ;
    float* o_sent   = o_tree + (size_t)TSZ * BSZ * S1SZ;

    auto conv = [&](const float* s, __bf16* d, size_t n) {
        int blocks = (int)((n + 255) / 256); if (blocks > 4096) blocks = 4096;
        k_f32_to_bf16<<<blocks, 256, 0, stream>>>(s, d, (int)n);
    };
    conv(Wih0, wih0b, (size_t)G3H * ESZ);
    conv(Whh0, whh0b, (size_t)G3H * HSZ);
    conv(Wih1, wih1b, (size_t)G3H * HSZ);
    conv(Whh1, whh1b, (size_t)G3H * HSZ);
    conv(tWin,  tWinb,  (size_t)HSZ * HSZ);
    conv(tWout, tWoutb, (size_t)HSZ * 2 * HSZ);
    conv(sWin,  sWinb,  (size_t)HSZ * HSZ);
    conv(sWout, sWoutb, (size_t)HSZ * 2 * HSZ);
    conv(outW,  outWb,  (size_t)VSZ * PSZ);

    k_init_hidden<<<(BSZ * HSZ + 255) / 256, 256, 0, stream>>>(h0f, h1f, h0b, h1b);

    const int bhBlocks = (BSZ * HSZ + 255) / 256;

    // GEMM grid: N / 256 blocks of 128 threads (4 waves x 4 N-tiles x 16)
    for (int t = 0; t < TSZ; ++t) {
        k_embed<<<(BSZ * ESZ + 255) / 256, 256, 0, stream>>>(seq, embTab, embb, t);

        // GRU layer 0
        k_gemm_bf16<<<G3H / 256, 128, 0, stream>>>(embb, ESZ, wih0b, ESZ, gi, G3H, nullptr, ESZ);
        k_gemm_bf16<<<G3H / 256, 128, 0, stream>>>(h0b, HSZ, whh0b, HSZ, gh, G3H, nullptr, HSZ);
        k_gru_gate<<<bhBlocks, 256, 0, stream>>>(gi, gh, bih0, bhh0, h0f, h0f, h0b);

        // GRU layer 1
        k_gemm_bf16<<<G3H / 256, 128, 0, stream>>>(h0b, HSZ, wih1b, HSZ, gi, G3H, nullptr, HSZ);
        k_gemm_bf16<<<G3H / 256, 128, 0, stream>>>(h1b, HSZ, whh1b, HSZ, gh, G3H, nullptr, HSZ);
        k_gru_gate<<<bhBlocks, 256, 0, stream>>>(gi, gh, bih1, bhh1, h1f, h1f, h1b);

        // tree attention (h = h1)
        k_gemm_bf16<<<HSZ / 256, 128, 0, stream>>>(h1b, HSZ, tWinb, HSZ, tq, HSZ, nullptr, HSZ);
        k_attn_scores<<<(BSZ * S1SZ * 32 + 127) / 128, 128, 0, stream>>>(ctx1, tq, sc, S1SZ);
        k_attn_softmax<<<BSZ, S1SZ, 0, stream>>>(sc, mask1, aws, o_tree + (size_t)t * BSZ * S1SZ, S1SZ);
        k_wctx_cat<<<bhBlocks, 256, 0, stream>>>(aws, ctx1, h1f, catb, S1SZ);
        k_gemm_bf16<<<HSZ / 256, 128, 0, stream>>>(catb, 2 * HSZ, tWoutb, 2 * HSZ, pre, HSZ, nullptr, 2 * HSZ);
        k_tanh<<<bhBlocks, 256, 0, stream>>>(pre, att1f, att1b);

        // sent attention (h = att1)
        k_gemm_bf16<<<HSZ / 256, 128, 0, stream>>>(att1b, HSZ, sWinb, HSZ, tq, HSZ, nullptr, HSZ);
        k_attn_scores<<<(BSZ * S2SZ * 32 + 127) / 128, 128, 0, stream>>>(ctx2, tq, sc, S2SZ);
        k_attn_softmax<<<BSZ, S2SZ, 0, stream>>>(sc, mask2, aws, o_sent + (size_t)t * BSZ * S2SZ, S2SZ);
        k_wctx_cat<<<bhBlocks, 256, 0, stream>>>(aws, ctx2, att1f, catb, S2SZ);
        k_gemm_bf16<<<HSZ / 256, 128, 0, stream>>>(catb, 2 * HSZ, sWoutb, 2 * HSZ, pre, HSZ, nullptr, 2 * HSZ);
        k_tanh<<<bhBlocks, 256, 0, stream>>>(pre, att2f, att2b);

        // output projection + log_softmax
        k_pred_cat<<<(BSZ * PSZ + 255) / 256, 256, 0, stream>>>(att2b, embb, predb);
        k_gemm_bf16<<<VSZ / 16 / 16, 128, 0, stream>>>(predb, PSZ, outWb, PSZ, logit, VSZ, outB, PSZ);
        k_log_softmax<<<BSZ, 256, 0, stream>>>(logit, o_scores + (size_t)t * BSZ * VSZ);
    }

    k_finalize<<<bhBlocks, 256, 0, stream>>>(h0f, h1f, att2f, o_hidden, o_output);
}
